// StarlingAttention_38027640439254
// MI455X (gfx1250) — matrix operand order repaired and checked by reading, not compiled, not run on hardware
//
#include <hip/hip_runtime.h>
#include <math.h>

// ---------------- problem constants (fixed by the reference) ----------------
#define NB     2
#define LQ     2048
#define DIM    1024
#define HEADS  16
#define HD     64
#define WIN    64
#define GM     (NB * LQ)          // 4096 rows in the big GEMMs
#define NEG_INF_F (-1e30f)

typedef _Float16 h16 __attribute__((ext_vector_type(16)));
typedef _Float16 h8  __attribute__((ext_vector_type(8)));
typedef _Float16 h4  __attribute__((ext_vector_type(4)));
typedef float    f8  __attribute__((ext_vector_type(8)));

static __device__ inline h16 cat8(h8 lo, h8 hi) {
  return __builtin_shufflevector(lo, hi, 0,1,2,3,4,5,6,7,8,9,10,11,12,13,14,15);
}

// A fragment, 16x32 f16 (M x K), row-major source with leading dim ld.
// ISA layout: lanes 0-15 hold K=[0..7] (v0-3) and K=[16..23] (v4-7) of row M=lane;
// lanes 16-31 hold K=[8..15] and K=[24..31] of row M=lane-16.
static __device__ inline h16 load_a_frag(const _Float16* base, int ld, int lane) {
  int row  = lane & 15;
  int koff = (lane >> 4) << 3;               // 0 or 8
  const _Float16* p = base + row * ld + koff;
  h8 lo = *(const h8*)(p);
  h8 hi = *(const h8*)(p + 16);
  return cat8(lo, hi);
}

// B fragment, 32x16 f16 (K x N), where B[k][n] = base[n*ld + k].
// ISA layout: lanes 0-15 hold K=[0..7] and K=[16..23] of column n; lanes 16-31 hold K=[8..15] and K=[24..31].
static __device__ inline h16 load_b_frag(const _Float16* base, int ld, int lane) {
  int n   = lane & 15;
  int klo = (lane >> 4) << 3;                // 0 or 8
  const _Float16* p = base + n * ld + klo;
  h8 lo = *(const h8*)(p);
  h8 hi = *(const h8*)(p + 16);
  return cat8(lo, hi);
}

static __device__ inline f8 wmma_f16(h16 a, h16 b, f8 c) {
  return __builtin_amdgcn_wmma_f32_16x16x32_f16(false, a, false, b, (short)0, c,
                                                false, false);
}

// ---------------- fp32 -> f16 convert (vectorized x4) ----------------
__global__ __launch_bounds__(256)
void cvt_f32_to_f16_kernel(const float* __restrict__ src,
                           _Float16* __restrict__ dst, int n4) {
  int i = blockIdx.x * blockDim.x + threadIdx.x;
  if (i < n4) {
    float4 v = reinterpret_cast<const float4*>(src)[i];
    h4 o = { (_Float16)v.x, (_Float16)v.y, (_Float16)v.z, (_Float16)v.w };
    reinterpret_cast<h4*>(dst)[i] = o;
  }
}

// ---------------- WMMA GEMM: C = A(4096x1024) @ W(1024x1024)^T ----------------
// mode 0: write f16 to [B,H,L,hd]   (Q)
// mode 1: write f16 to [B,H,L,hd]   (K)
// mode 2: write f16 to [B,H,hd,L]   (V transposed)
// mode 3: write f32 to [B,L,D]      (final output)
__global__ __launch_bounds__(256)
void gemm_wmma_kernel(const _Float16* __restrict__ A,
                      const _Float16* __restrict__ W,
                      _Float16* __restrict__ outh,
                      float* __restrict__ outf, int mode) {
  const int K = DIM;
  int lane = threadIdx.x & 31;
  int w    = threadIdx.x >> 5;               // 8 waves
  int m0 = blockIdx.y * 64 + (w >> 2) * 32;  // 2 waves along M
  int n0 = blockIdx.x * 128 + (w & 3) * 32;  // 4 waves along N

  f8 c00 = {}, c01 = {}, c10 = {}, c11 = {};
  const _Float16* arow0 = A + (size_t)m0 * K;
  const _Float16* arow1 = A + (size_t)(m0 + 16) * K;
  const _Float16* brow0 = W + (size_t)n0 * K;
  const _Float16* brow1 = W + (size_t)(n0 + 16) * K;

  for (int k = 0; k < K; k += 32) {
    if (k + 64 < K) {
      __builtin_prefetch(arow0 + (lane & 15) * K + k + 64, 0, 1);
      __builtin_prefetch(brow0 + (lane & 15) * K + k + 64, 0, 1);
    }
    h16 a0 = load_a_frag(arow0 + k, K, lane);
    h16 a1 = load_a_frag(arow1 + k, K, lane);
    h16 b0 = load_b_frag(brow0 + k, K, lane);
    h16 b1 = load_b_frag(brow1 + k, K, lane);
    c00 = wmma_f16(a0, b0, c00);
    c01 = wmma_f16(a0, b1, c01);
    c10 = wmma_f16(a1, b0, c10);
    c11 = wmma_f16(a1, b1, c11);
  }

  int hi8 = ((lane >> 4) & 1) * 8;
  int nc  = lane & 15;
  f8 acc[2][2] = {{c00, c01}, {c10, c11}};
  #pragma unroll
  for (int ti = 0; ti < 2; ++ti) {
    #pragma unroll
    for (int tj = 0; tj < 2; ++tj) {
      #pragma unroll
      for (int r = 0; r < 8; ++r) {
        int row = m0 + ti * 16 + r + hi8;
        int col = n0 + tj * 16 + nc;
        float v = acc[ti][tj][r];
        if (mode == 3) {
          outf[(size_t)row * DIM + col] = v;
        } else {
          _Float16 hv = (_Float16)v;
          int b = row >> 11, l = row & (LQ - 1);
          int h = col >> 6,  d = col & (HD - 1);
          size_t bh = (size_t)(b * HEADS + h);
          if (mode == 2) outh[(bh * HD + d) * LQ + l] = hv;   // V^T
          else           outh[(bh * LQ + l) * HD + d] = hv;   // Q / K
        }
      }
    }
  }
}

// ---------------- kmean[b,h,d] = mean_l K[b,h,l,d] ----------------
__global__ __launch_bounds__(256)
void kmean_kernel(const _Float16* __restrict__ Kh, float* __restrict__ kmean) {
  int idx = blockIdx.x * blockDim.x + threadIdx.x;   // 0..NB*HEADS*HD-1
  int d  = idx & (HD - 1);
  int bh = idx >> 6;
  const _Float16* p = Kh + (size_t)bh * LQ * HD + d;
  float s = 0.f;
  for (int l = 0; l < LQ; ++l) s += (float)p[(size_t)l * HD];
  kmean[idx] = s * (1.0f / LQ);
}

// ---------------- rowmean[b,h,l] = dot(Q[b,h,l,:], kmean[b,h,:]) / sqrt(hd) --
__global__ __launch_bounds__(256)
void rowmean_kernel(const _Float16* __restrict__ Qh,
                    const float* __restrict__ kmean,
                    float* __restrict__ rowmean) {
  int idx = blockIdx.x * blockDim.x + threadIdx.x;   // 0..NB*HEADS*LQ-1
  int l  = idx & (LQ - 1);
  int bh = idx >> 11;
  const _Float16* q = Qh + ((size_t)bh * LQ + l) * HD;
  const float* km = kmean + bh * HD;
  float s = 0.f;
  #pragma unroll 8
  for (int d = 0; d < HD; ++d) s += (float)q[d] * km[d];
  rowmean[idx] = s * 0.125f;                         // / sqrt(64)
}

// ---------------- sliding-window attention: one wave per 16-query block ------
__global__ __launch_bounds__(128)
void attn_wmma_kernel(const _Float16* __restrict__ Qh,
                      const _Float16* __restrict__ Kh,
                      const _Float16* __restrict__ Vth,
                      const float* __restrict__ rowmean,
                      const float* __restrict__ sep_g,
                      const float* __restrict__ ali_g,
                      const float* __restrict__ coh_g,
                      _Float16* __restrict__ attnout) {
  __shared__ _Float16 pLDS[4][16][96];               // wave-private P tiles
  int lane = threadIdx.x & 31;
  int w    = threadIdx.x >> 5;
  int gid  = blockIdx.x * 4 + w;                     // 0..4095
  int qb = gid & 127;                                // L/16 = 128 blocks
  int h  = (gid >> 7) & (HEADS - 1);
  int b  = gid >> 11;
  int q0 = qb * 16;

  const _Float16* Qb = Qh  + (size_t)(b * HEADS + h) * LQ * HD;
  const _Float16* Kb = Kh  + (size_t)(b * HEADS + h) * LQ * HD;
  const _Float16* Vb = Vth + (size_t)(b * HEADS + h) * HD * LQ;

  float sep = *sep_g, ali = *ali_g, coh = *coh_g;
  int hi8 = ((lane >> 4) & 1) * 8;
  int nc  = lane & 15;

  // Q fragments (K-dims 0..31 and 32..63)
  h16 a0 = load_a_frag(Qb + (size_t)q0 * HD +  0, HD, lane);
  h16 a1 = load_a_frag(Qb + (size_t)q0 * HD + 32, HD, lane);

  // per-lane row means (row = q0 + r + hi8)
  float rm[8];
  const float* rmb = rowmean + (size_t)(b * HEADS + h) * LQ + q0;
  #pragma unroll
  for (int r = 0; r < 8; ++r) rm[r] = rmb[r + hi8];

  // ---- scores for 5 key tiles covering [q0-64, q0+15] ----
  float sv[5][8];
  #pragma unroll
  for (int t = 0; t < 5; ++t) {
    int jt = q0 - 64 + t * 16;
    if (jt < 0) {                                    // uniform per wave
      #pragma unroll
      for (int r = 0; r < 8; ++r) sv[t][r] = NEG_INF_F;
      continue;
    }
    h16 kb0 = load_b_frag(Kb + (size_t)jt * HD +  0, HD, lane);
    h16 kb1 = load_b_frag(Kb + (size_t)jt * HD + 32, HD, lane);
    f8 s = {};
    s = wmma_f16(a0, kb0, s);
    s = wmma_f16(a1, kb1, s);
    #pragma unroll
    for (int r = 0; r < 8; ++r) {
      float sc  = s[r] * 0.125f;                     // / sqrt(hd)
      float sig = 1.0f / (1.0f + __expf(-sc));       // sigmoid(score)
      float val = sc + ali * sc - sep * sig * sig - coh * fabsf(sc - rm[r]);
      int dist = (q0 + r + hi8) - (jt + nc);
      sv[t][r] = (dist < 0 || dist >= WIN) ? NEG_INF_F : val;
    }
  }

  // ---- row softmax (rows live in 16-lane groups; xor<16 keeps the group) ----
  float inv_sum[8];
  #pragma unroll
  for (int r = 0; r < 8; ++r) {
    float m = sv[0][r];
    #pragma unroll
    for (int t = 1; t < 5; ++t) m = fmaxf(m, sv[t][r]);
    #pragma unroll
    for (int off = 8; off >= 1; off >>= 1) m = fmaxf(m, __shfl_xor(m, off, 32));
    float ssum = 0.f;
    #pragma unroll
    for (int t = 0; t < 5; ++t) {
      float e = __expf(sv[t][r] - m);
      sv[t][r] = e;
      ssum += e;
    }
    #pragma unroll
    for (int off = 8; off >= 1; off >>= 1) ssum += __shfl_xor(ssum, off, 32);
    inv_sum[r] = 1.0f / ssum;
  }

  // ---- C-layout -> A-layout via LDS round trip ----
  #pragma unroll
  for (int t = 0; t < 5; ++t)
    #pragma unroll
    for (int r = 0; r < 8; ++r)
      pLDS[w][r + hi8][t * 16 + nc] = (_Float16)sv[t][r];
  for (int idx = lane; idx < 256; idx += 32)         // zero pad cols 80..95
    pLDS[w][idx >> 4][80 + (idx & 15)] = (_Float16)0.f;
  __syncthreads();

  h16 p0 = load_a_frag(&pLDS[w][0][0] +  0, 96, lane);
  h16 p1 = load_a_frag(&pLDS[w][0][0] + 32, 96, lane);
  h16 p2 = load_a_frag(&pLDS[w][0][0] + 64, 96, lane);
  h16 pfrag[3] = {p0, p1, p2};

  // ---- O = P(16x96) @ V(96x64); V^T rows are contiguous in l ----
  f8 o[4] = {};
  #pragma unroll
  for (int f = 0; f < 3; ++f) {
    int jbase = q0 - 64 + f * 32;
    #pragma unroll
    for (int dt = 0; dt < 4; ++dt) {
      // custom B fragment with clamped (garbage-safe, p==0 there) column range
      int klo = (lane >> 4) << 3;
      int off = jbase + klo;
      off = off < 0 ? 0 : (off > (LQ - 24) ? (LQ - 24) : off);
      const _Float16* vp = Vb + (size_t)(dt * 16 + nc) * LQ + off;
      h8 lo = *(const h8*)(vp);
      h8 hi = *(const h8*)(vp + 16);
      h16 bv = cat8(lo, hi);
      o[dt] = wmma_f16(pfrag[f], bv, o[dt]);
    }
  }

  // ---- normalize and scatter to [B, L, D] (head-interleaved) as f16 ----
  #pragma unroll
  for (int dt = 0; dt < 4; ++dt)
    #pragma unroll
    for (int r = 0; r < 8; ++r) {
      float val = o[dt][r] * inv_sum[r];
      int i   = q0 + r + hi8;
      int col = h * HD + dt * 16 + nc;
      attnout[((size_t)b * LQ + i) * DIM + col] = (_Float16)val;
    }
}

// ---------------------------------------------------------------------------
extern "C" void kernel_launch(void* const* d_in, const int* in_sizes, int n_in,
                              void* d_out, int out_size, void* d_ws, size_t ws_size,
                              hipStream_t stream) {
  const float* x     = (const float*)d_in[0];
  const float* Wq    = (const float*)d_in[1];
  const float* Wk    = (const float*)d_in[2];
  const float* Wv    = (const float*)d_in[3];
  const float* Wo    = (const float*)d_in[4];
  const float* sep_g = (const float*)d_in[5];
  const float* ali_g = (const float*)d_in[6];
  const float* coh_g = (const float*)d_in[7];
  float* out = (float*)d_out;

  // workspace carve-up (all sizes are multiples of 256B)
  char* p = (char*)d_ws;
  _Float16* xh  = (_Float16*)p; p += (size_t)GM  * DIM * 2;
  _Float16* wqh = (_Float16*)p; p += (size_t)DIM * DIM * 2;
  _Float16* wkh = (_Float16*)p; p += (size_t)DIM * DIM * 2;
  _Float16* wvh = (_Float16*)p; p += (size_t)DIM * DIM * 2;
  _Float16* woh = (_Float16*)p; p += (size_t)DIM * DIM * 2;
  _Float16* Qh  = (_Float16*)p; p += (size_t)GM  * DIM * 2;
  _Float16* Kh  = (_Float16*)p; p += (size_t)GM  * DIM * 2;
  _Float16* Vth = (_Float16*)p; p += (size_t)GM  * DIM * 2;
  _Float16* AOh = (_Float16*)p; p += (size_t)GM  * DIM * 2;
  float* kmean  = (float*)p;    p += (size_t)NB * HEADS * HD * 4;
  float* rmean  = (float*)p;    p += (size_t)NB * HEADS * LQ * 4;

  // 1) fp32 -> f16 converts
  cvt_f32_to_f16_kernel<<<(GM * DIM / 4) / 256, 256, 0, stream>>>(x, xh, GM * DIM / 4);
  cvt_f32_to_f16_kernel<<<(DIM * DIM / 4) / 256, 256, 0, stream>>>(Wq, wqh, DIM * DIM / 4);
  cvt_f32_to_f16_kernel<<<(DIM * DIM / 4) / 256, 256, 0, stream>>>(Wk, wkh, DIM * DIM / 4);
  cvt_f32_to_f16_kernel<<<(DIM * DIM / 4) / 256, 256, 0, stream>>>(Wv, wvh, DIM * DIM / 4);
  cvt_f32_to_f16_kernel<<<(DIM * DIM / 4) / 256, 256, 0, stream>>>(Wo, woh, DIM * DIM / 4);

  // 2) Q/K/V projections (WMMA GEMMs)
  dim3 gg(DIM / 128, GM / 64);
  gemm_wmma_kernel<<<gg, 256, 0, stream>>>(xh, wqh, Qh,  nullptr, 0);
  gemm_wmma_kernel<<<gg, 256, 0, stream>>>(xh, wkh, Kh,  nullptr, 1);
  gemm_wmma_kernel<<<gg, 256, 0, stream>>>(xh, wvh, Vth, nullptr, 2);

  // 3) cohesion helpers: key mean, then per-row score mean
  kmean_kernel<<<(NB * HEADS * HD) / 256, 256, 0, stream>>>(Kh, kmean);
  rowmean_kernel<<<(NB * HEADS * LQ) / 256, 256, 0, stream>>>(Qh, kmean, rmean);

  // 4) windowed attention (one wave per 16-query block)
  attn_wmma_kernel<<<(NB * HEADS * (LQ / 16)) / 4, 128, 0, stream>>>(
      Qh, Kh, Vth, rmean, sep_g, ali_g, coh_g, AOh);

  // 5) output projection, f32 result
  gemm_wmma_kernel<<<gg, 256, 0, stream>>>(AOh, woh, nullptr, out, 3);
}
